// AttentionLayer_10230612099706
// MI455X (gfx1250) — matrix operand order repaired and checked by reading, compiled, not run on hardware
//
#include <hip/hip_runtime.h>

// Problem constants (fixed by the reference)
#define BB 4
#define SS 2048
#define HH 16
#define DHH 64
#define DMM 1024

// Flip to 0 if the async-to-LDS inline asm is rejected by the assembler.
#ifndef USE_ASYNC_STAGE
#define USE_ASYNC_STAGE 1
#endif

typedef __attribute__((ext_vector_type(16))) __bf16 v16bf;
typedef __attribute__((ext_vector_type(8)))  float  v8f;

union BFrag {
    v16bf v;
    uint4 q[2];
    unsigned int u[8];
    unsigned short s[16];
};
union AccF {
    v8f v;
    float f[8];
};

__device__ __forceinline__ unsigned short f2bf(float x) {
    unsigned u = __float_as_uint(x);
    u += 0x7FFFu + ((u >> 16) & 1u);   // round-to-nearest-even
    return (unsigned short)(u >> 16);
}

__device__ __forceinline__ v8f wmma_bf16(const BFrag& a, const BFrag& b, v8f c) {
    // (neg_a, A, neg_b, B, c_mod, C, reuse_a, reuse_b)
    return __builtin_amdgcn_wmma_f32_16x16x32_bf16(false, a.v, false, b.v,
                                                   (short)0, c, false, false);
}

// Generic pointer -> LDS byte offset (low 32 bits of the generic address).
__device__ __forceinline__ unsigned lds_addr(const void* p) {
    return (unsigned)(unsigned long long)p;
}

// ---------------------------------------------------------------------------
// QKV projection GEMM: out = X(8192x1024) @ W(1024x1024) + bias, emitted bf16.
// v_transpose==0: out layout (b,h,t,d)   [for Q and K]
// v_transpose==1: out layout (b,h,d,t)   [for V, pre-transposed for P.V]
// Block = 256 threads = 8 waves; block tile 64(M) x 128(N); wave tile 16x64.
// ---------------------------------------------------------------------------
__global__ __launch_bounds__(256) void qkv_gemm_kernel(
    const float* __restrict__ X, const float* __restrict__ W,
    const float* __restrict__ bias, unsigned short* __restrict__ out,
    int v_transpose)
{
    __shared__ unsigned short As[64][32];    // [m][k] bf16
    __shared__ unsigned short Bsh[128][32];  // [n][k] bf16 (transposed W tile)

    const int tid  = threadIdx.x;
    const int wave = tid >> 5, lane = tid & 31;
    const int lm = lane & 15, lg = lane >> 4;
    const int m0 = blockIdx.y * 64;
    const int n0 = blockIdx.x * 128;
    const int wm = wave >> 1, wn = wave & 1;

    AccF acc[4];
    #pragma unroll
    for (int i = 0; i < 4; ++i)
        #pragma unroll
        for (int j = 0; j < 8; ++j) acc[i].f[j] = 0.0f;

    for (int k0 = 0; k0 < DMM; k0 += 32) {
        // --- stage A tile 64x32 (f32 -> bf16), 8 elements per thread
        {
            int row = tid >> 2;
            int col = (tid & 3) * 8;
            const float4* src = (const float4*)(X + (size_t)(m0 + row) * DMM + k0 + col);
            float4 x0 = src[0], x1 = src[1];
            uint4 p;
            p.x = (unsigned)f2bf(x0.x) | ((unsigned)f2bf(x0.y) << 16);
            p.y = (unsigned)f2bf(x0.z) | ((unsigned)f2bf(x0.w) << 16);
            p.z = (unsigned)f2bf(x1.x) | ((unsigned)f2bf(x1.y) << 16);
            p.w = (unsigned)f2bf(x1.z) | ((unsigned)f2bf(x1.w) << 16);
            *(uint4*)&As[row][col] = p;
        }
        // --- stage B tile 32x128 transposed -> Bsh[n][k], 16 elems per thread
        {
            int c  = tid >> 3;          // k within tile: 0..31
            int jg = (tid & 7) * 16;    // n within tile: 0..112
            const float4* src = (const float4*)(W + (size_t)(k0 + c) * DMM + n0 + jg);
            #pragma unroll
            for (int v4 = 0; v4 < 4; ++v4) {
                float4 w4 = src[v4];
                Bsh[jg + v4*4 + 0][c] = f2bf(w4.x);
                Bsh[jg + v4*4 + 1][c] = f2bf(w4.y);
                Bsh[jg + v4*4 + 2][c] = f2bf(w4.z);
                Bsh[jg + v4*4 + 3][c] = f2bf(w4.w);
            }
        }
        __syncthreads();

        // A fragment: lane = row (wm*16+lm); K pairs: [khalf..khalf+7] and [16+khalf..+7]
        BFrag a;
        {
            const uint4* ar = (const uint4*)&As[wm*16 + lm][0];
            a.q[0] = ar[lg];        // k = lg*8 .. lg*8+7
            a.q[1] = ar[2 + lg];    // k = 16+lg*8 .. +7
        }
        #pragma unroll
        for (int nb = 0; nb < 4; ++nb) {
            BFrag bf;
            const uint4* br = (const uint4*)&Bsh[wn*64 + nb*16 + lm][0];
            bf.q[0] = br[lg];       // k = lg*8 .. +7
            bf.q[1] = br[2 + lg];   // k = 16+lg*8 .. +7
            acc[nb].v = wmma_bf16(a, bf, acc[nb].v);
        }
        __syncthreads();
    }

    // --- epilogue: bias, bf16 convert, layout-specific store
    #pragma unroll
    for (int nb = 0; nb < 4; ++nb) {
        int ng = n0 + wn*64 + nb*16 + lm;       // global column
        float bv = bias[ng];
        int head = ng >> 6, d = ng & 63;
        if (!v_transpose) {
            #pragma unroll
            for (int j = 0; j < 8; ++j) {
                int mg = m0 + wm*16 + lg*8 + j; // global row (token)
                int bb = mg >> 11, ff = mg & 2047;
                out[(((size_t)(bb*HH + head)) * SS + ff) * DHH + d] =
                    f2bf(acc[nb].f[j] + bv);
            }
        } else {
            int fbase = m0 + wm*16 + lg*8;
            int bb = fbase >> 11, ff = fbase & 2047;
            unsigned short h8[8];
            #pragma unroll
            for (int j = 0; j < 8; ++j) h8[j] = f2bf(acc[nb].f[j] + bv);
            uint4 p;
            p.x = (unsigned)h8[0] | ((unsigned)h8[1] << 16);
            p.y = (unsigned)h8[2] | ((unsigned)h8[3] << 16);
            p.z = (unsigned)h8[4] | ((unsigned)h8[5] << 16);
            p.w = (unsigned)h8[6] | ((unsigned)h8[7] << 16);
            *(uint4*)&out[(((size_t)(bb*HH + head)) * DHH + d) * SS + ff] = p;
        }
    }
}

// ---------------------------------------------------------------------------
// Flash attention: one wave per (b, h, 16-query tile); all 8 waves of a block
// share the same (b,h), so K/Vt tiles (32 keys) are staged into LDS once per
// block per iteration — via global_load_async_to_lds_b128 (ASYNCcnt) when
// available. Q (b,h,t,d), K (b,h,t,d), Vt (b,h,d,t), all bf16.
// ---------------------------------------------------------------------------
__global__ __launch_bounds__(256) void flash_attn_kernel(
    const unsigned short* __restrict__ Q, const unsigned short* __restrict__ K,
    const unsigned short* __restrict__ Vt, const int* __restrict__ to_mask,
    float* __restrict__ out)
{
    __shared__ unsigned short Ks[32 * 64];      // [t_local][d]      4 KB
    __shared__ unsigned short Vs[64 * 32];      // [d][t_local]      4 KB
    __shared__ unsigned short Ps[8][16 * 32];   // per-wave P tiles  8 KB

    const int tid  = threadIdx.x;
    const int wave = tid >> 5, lane = tid & 31;
    const int lm = lane & 15, lg = lane >> 4;
    const int gid = blockIdx.x * 8 + wave;
    const int qt = gid & 127;           // 128 query tiles per (b,h)
    const int bh = gid >> 7;            // uniform across the block
    const int b = bh >> 4, h = bh & 15;
    const int m0 = qt * 16;

    const unsigned short* Qp = Q  + ((size_t)bh * SS + m0) * DHH;
    const unsigned short* Kp = K  + (size_t)bh * SS * DHH;
    const unsigned short* Vp = Vt + (size_t)bh * DHH * SS;
    const int* tm = to_mask + b * SS;

    // Q A-fragments for d=[0,32) and d=[32,64)
    BFrag a0, a1;
    {
        const uint4* qr = (const uint4*)(Qp + (size_t)lm * DHH);
        a0.q[0] = qr[2*lg];     a0.q[1] = qr[2*lg + 1];
        a1.q[0] = qr[4 + 2*lg]; a1.q[1] = qr[5 + 2*lg];
    }

    AccF o[4];
    #pragma unroll
    for (int i = 0; i < 4; ++i)
        #pragma unroll
        for (int j = 0; j < 8; ++j) o[i].f[j] = 0.0f;
    float mrow[8], lrow[8];
    #pragma unroll
    for (int j = 0; j < 8; ++j) { mrow[j] = -1e30f; lrow[j] = 0.0f; }

#if USE_ASYNC_STAGE
    const unsigned ks_lds = lds_addr(&Ks[0]);
    const unsigned vs_lds = lds_addr(&Vs[0]);
#endif
    // Per-thread staging slots: K tile = 512 uint4 chunks (2/thread),
    // V tile = 256 uint4 chunks (1/thread).
    const int c0 = tid, c1 = tid + 256;

    for (int t0 = 0; t0 < SS; t0 += 32) {
        // --- cooperative staging of K (32x64) and Vt (64x32) bf16 tiles
        {
            const unsigned short* ksrc0 = Kp + (size_t)(t0 + (c0 >> 3)) * DHH + (c0 & 7) * 8;
            const unsigned short* ksrc1 = Kp + (size_t)(t0 + (c1 >> 3)) * DHH + (c1 & 7) * 8;
            const unsigned short* vsrc  = Vp + (size_t)(tid >> 2) * SS + t0 + (tid & 3) * 8;
#if USE_ASYNC_STAGE
            asm volatile("global_load_async_to_lds_b128 %0, %1, off"
                         :: "v"(ks_lds + (unsigned)c0 * 16u),
                            "v"((unsigned long long)(size_t)ksrc0) : "memory");
            asm volatile("global_load_async_to_lds_b128 %0, %1, off"
                         :: "v"(ks_lds + (unsigned)c1 * 16u),
                            "v"((unsigned long long)(size_t)ksrc1) : "memory");
            asm volatile("global_load_async_to_lds_b128 %0, %1, off"
                         :: "v"(vs_lds + (unsigned)tid * 16u),
                            "v"((unsigned long long)(size_t)vsrc) : "memory");
            asm volatile("s_wait_asynccnt 0" ::: "memory");
#else
            uint4 kv0 = *(const uint4*)ksrc0;
            uint4 kv1 = *(const uint4*)ksrc1;
            uint4 vv  = *(const uint4*)vsrc;
            *(uint4*)&Ks[c0 * 8]  = kv0;
            *(uint4*)&Ks[c1 * 8]  = kv1;
            *(uint4*)&Vs[tid * 8] = vv;
#endif
            __syncthreads();
        }

        // --- scores: two 16x16 tiles (cols t0..t0+15 and t0+16..t0+31)
        AccF s0, s1;
        #pragma unroll
        for (int j = 0; j < 8; ++j) { s0.f[j] = 0.0f; s1.f[j] = 0.0f; }
        {
            const uint4* kr = (const uint4*)&Ks[lm * 64];
            BFrag kb;
            kb.q[0] = kr[2*lg];     kb.q[1] = kr[2*lg + 1];   // d 0..31
            s0.v = wmma_bf16(a0, kb, s0.v);
            kb.q[0] = kr[4 + 2*lg]; kb.q[1] = kr[5 + 2*lg];   // d 32..63
            s0.v = wmma_bf16(a1, kb, s0.v);
        }
        {
            const uint4* kr = (const uint4*)&Ks[(16 + lm) * 64];
            BFrag kb;
            kb.q[0] = kr[2*lg];     kb.q[1] = kr[2*lg + 1];
            s1.v = wmma_bf16(a0, kb, s1.v);
            kb.q[0] = kr[4 + 2*lg]; kb.q[1] = kr[5 + 2*lg];
            s1.v = wmma_bf16(a1, kb, s1.v);
        }

        const float add0 = (float)(1 - tm[t0 + lm])      * -10000.0f;
        const float add1 = (float)(1 - tm[t0 + 16 + lm]) * -10000.0f;

        // --- online softmax update (rows = j + 8*lg, 16-lane butterflies)
        float p0[8], p1[8], alpha[8];
        #pragma unroll
        for (int j = 0; j < 8; ++j) {
            float v0 = s0.f[j] * 0.125f + add0;
            float v1 = s1.f[j] * 0.125f + add1;
            float r = fmaxf(v0, v1);
            #pragma unroll
            for (int off = 8; off >= 1; off >>= 1)
                r = fmaxf(r, __shfl_xor(r, off, 32));
            float mn = fmaxf(mrow[j], r);
            float e0 = __expf(v0 - mn);
            float e1 = __expf(v1 - mn);
            float rs = e0 + e1;
            #pragma unroll
            for (int off = 8; off >= 1; off >>= 1)
                rs += __shfl_xor(rs, off, 32);
            alpha[j] = __expf(mrow[j] - mn);
            lrow[j]  = lrow[j] * alpha[j] + rs;
            mrow[j]  = mn;
            p0[j] = e0; p1[j] = e1;
        }
        #pragma unroll
        for (int nb = 0; nb < 4; ++nb)
            #pragma unroll
            for (int j = 0; j < 8; ++j) o[nb].f[j] *= alpha[j];

        // --- re-layout P (C layout) -> A operand via per-wave LDS tile
        unsigned short* pw = &Ps[wave][0];
        #pragma unroll
        for (int j = 0; j < 8; ++j) {
            pw[(lg*8 + j)*32 + lm]      = f2bf(p0[j]);
            pw[(lg*8 + j)*32 + 16 + lm] = f2bf(p1[j]);
        }
        BFrag pa;
        {
            const unsigned* pr = (const unsigned*)&Ps[wave][lm * 32];
            #pragma unroll
            for (int i = 0; i < 4; ++i) {
                pa.u[i]     = pr[lg*4 + i];        // k = lg*8 + 2i
                pa.u[4 + i] = pr[8 + lg*4 + i];    // k = 16 + lg*8 + 2i
            }
        }

        // --- O += P * V  (B operand from staged Vt tile: contiguous in t)
        #pragma unroll
        for (int nb = 0; nb < 4; ++nb) {
            const uint4* vr = (const uint4*)&Vs[(nb*16 + lm) * 32];
            BFrag vb;
            vb.q[0] = vr[lg];
            vb.q[1] = vr[2 + lg];
            o[nb].v = wmma_bf16(pa, vb, o[nb].v);
        }

        __syncthreads();   // protect K/V tiles before next staging round
    }

    // --- finalize: divide by softmax denom, store f32 (b, f, h*64+d)
    float inv[8];
    #pragma unroll
    for (int j = 0; j < 8; ++j) inv[j] = 1.0f / lrow[j];
    #pragma unroll
    for (int nb = 0; nb < 4; ++nb) {
        int col = h*DHH + nb*16 + lm;
        #pragma unroll
        for (int j = 0; j < 8; ++j) {
            int mg = m0 + lg*8 + j;
            out[((size_t)b * SS + mg) * (HH*DHH) + col] = o[nb].f[j] * inv[j];
        }
    }
}

// ---------------------------------------------------------------------------
extern "C" void kernel_launch(void* const* d_in, const int* in_sizes, int n_in,
                              void* d_out, int out_size, void* d_ws, size_t ws_size,
                              hipStream_t stream)
{
    (void)in_sizes; (void)n_in; (void)out_size; (void)ws_size;
    const float* from_t  = (const float*)d_in[0];
    const float* to_t    = (const float*)d_in[1];
    // d_in[2] = from_mask (unused by the reference output)
    const int*   to_mask = (const int*)d_in[3];
    const float* Wq = (const float*)d_in[4];
    const float* bq = (const float*)d_in[5];
    const float* Wk = (const float*)d_in[6];
    const float* bk = (const float*)d_in[7];
    const float* Wv = (const float*)d_in[8];
    const float* bv = (const float*)d_in[9];
    float* out = (float*)d_out;

    // Workspace: bf16 Q (b,h,t,d), K (b,h,t,d), Vt (b,h,d,t) — 16 MB each.
    const size_t elems = (size_t)BB * HH * SS * DHH;   // 8,388,608
    unsigned short* qb = (unsigned short*)d_ws;
    unsigned short* kb = qb + elems;
    unsigned short* vb = kb + elems;

    dim3 grid(DMM / 128, (BB * SS) / 64);  // (8, 128)
    qkv_gemm_kernel<<<grid, 256, 0, stream>>>(from_t, Wq, bq, qb, 0);
    qkv_gemm_kernel<<<grid, 256, 0, stream>>>(to_t,   Wk, bk, kb, 0);
    qkv_gemm_kernel<<<grid, 256, 0, stream>>>(to_t,   Wv, bv, vb, 1);

    flash_attn_kernel<<<(BB * HH * (SS/16)) / 8, 256, 0, stream>>>(
        qb, kb, vb, to_mask, out);
}
